// QLSTM_22771916603951
// MI455X (gfx1250) — matrix operand order repaired and checked by reading, not compiled, not run on hardware
//
#include <hip/hip_runtime.h>

// ---------------------------------------------------------------------------
// QLSTM for MI455X (gfx1250): bf16 WMMA (v_wmma_f32_16x16x32_bf16) for both
// the big input-projection GEMM and the sequential recurrent scan.
// B=16 == WMMA M-tile, so every timestep is exactly one WMMA row tile.
// ---------------------------------------------------------------------------

typedef __bf16 bf16_t;
typedef __attribute__((ext_vector_type(16))) __bf16 v16bf;
typedef __attribute__((ext_vector_type(8)))  __bf16 v8bf;
typedef __attribute__((ext_vector_type(8)))  float  v8f;

#define T_STEPS 1500
#define BATCH   16
#define DIM     1024
#define HID     1024
#define NBLK_SCAN 64
static const size_t GS = (size_t)T_STEPS * BATCH * HID;   // per-gate stride in G

// Hamilton product block structure of qmat():
//   [[ r,-i,-j,-k],[ i, r,-k, j],[ j, k, r,-i],[ k,-j, i, r]]
__constant__ int   qcomp[4][4] = {{0,1,2,3},{1,0,3,2},{2,3,0,1},{3,2,1,0}};
__constant__ float qsign[4][4] = {{ 1.f,-1.f,-1.f,-1.f},
                                  { 1.f, 1.f,-1.f, 1.f},
                                  { 1.f, 1.f, 1.f,-1.f},
                                  { 1.f,-1.f, 1.f, 1.f}};

// ---- WMMA fragment loaders (contiguous k = 16*hi + e in both A and B, wave32) ----------
// A (16x32 bf16, row-major source, leading dim ldk):
//   lane m (lanes 0-15) / lane 16+m hold row m; elems e = K 16*hi + e
//   -> two contiguous 16B loads.
__device__ __forceinline__ v16bf load_a_frag(const bf16_t* a, int ldk, int k0, int lane) {
  int m = lane & 15, hi = lane >> 4;
  const bf16_t* p = a + (size_t)m * ldk + k0 + 16 * hi;
  v8bf lo = *(const v8bf*)p;
  v8bf hf = *(const v8bf*)(p + 8);
  return __builtin_shufflevector(lo, hf, 0,1,2,3,4,5,6,7,8,9,10,11,12,13,14,15);
}

// B (32x16 bf16): lane holds column n = lane&15; elems e = K 16*hi + e.
// Source is stored COLUMN-major ([col][k], leading dim ldk) so this is two
// contiguous 16B loads per lane.
__device__ __forceinline__ v16bf load_b_frag(const bf16_t* colmaj, int ldk,
                                             int col0, int k0, int lane) {
  int n = lane & 15, hi = lane >> 4;
  const bf16_t* p = colmaj + (size_t)(col0 + n) * ldk + k0 + 16 * hi;
  v8bf b0 = *(const v8bf*)p;
  v8bf b1 = *(const v8bf*)(p + 8);
  return __builtin_shufflevector(b0, b1, 0,1,2,3,4,5,6,7,8,9,10,11,12,13,14,15);
}

__device__ __forceinline__ float fsigmoid(float x) { return 1.f / (1.f + __expf(-x)); }
__device__ __forceinline__ float ftanh(float x)    { return 2.f / (1.f + __expf(-2.f * x)) - 1.f; }

// ---- init: zero h double-buffer + barrier state ----------------------------
__global__ void init_kernel(bf16_t* hbuf, unsigned* bar) {
  int i = blockIdx.x * blockDim.x + threadIdx.x;
  if (i < 2 * BATCH * HID) hbuf[i] = (bf16_t)0.f;
  if (i < 64) bar[i] = 0u;
}

// ---- fp32 -> bf16 conversion ----------------------------------------------
__global__ void cvt_bf16_kernel(const float* __restrict__ src, bf16_t* __restrict__ dst, int n) {
  int i = blockIdx.x * blockDim.x + threadIdx.x;
  if (i < n) dst[i] = (bf16_t)src[i];
}

// ---- expand quaternion components into transposed (col-major) qmat, bf16 ---
// qt[col*1024 + k] = qmat[k][col]
__global__ void qexpand_kernel(const float* __restrict__ W4, bf16_t* __restrict__ qt) {
  int i = blockIdx.x * blockDim.x + threadIdx.x;   // i = col*1024 + k
  int col = i >> 10, k = i & 1023;
  int br = k >> 8, bc = col >> 8;
  float v = qsign[br][bc] *
            W4[(size_t)qcomp[br][bc] * 65536 + (size_t)(k & 255) * 256 + (col & 255)];
  qt[i] = (bf16_t)v;
}

// ---- phase 1: G[gate][t][b][h] = x @ qmat(W[gate]) + bias ------------------
// grid = (1500 M-tiles, 64 col-groups), block = 128 (4 waves, 1 N-tile/wave)
__global__ void gemm_gates_kernel(const bf16_t* __restrict__ xb,
                                  const bf16_t* __restrict__ qtw,
                                  const float* __restrict__ b0, const float* __restrict__ b1,
                                  const float* __restrict__ b2, const float* __restrict__ b3,
                                  float* __restrict__ G) {
  int lane = threadIdx.x & 31;
  int wave = threadIdx.x >> 5;
  int t = blockIdx.x;
  int ntile = blockIdx.y * 4 + wave;   // 0..255 over 4096 concatenated gate cols
  int colg = ntile << 4;
  int gate = colg >> 10;
  int col  = colg & 1023;

  const bf16_t* A  = xb  + (size_t)t * (BATCH * DIM);
  const bf16_t* Bm = qtw + ((size_t)gate << 20);

  v8f acc = {};
#pragma unroll 4
  for (int k0 = 0; k0 < DIM; k0 += 32) {
    v16bf a = load_a_frag(A, DIM, k0, lane);
    v16bf b = load_b_frag(Bm, DIM, col, k0, lane);
    acc = __builtin_amdgcn_wmma_f32_16x16x32_bf16(false, a, false, b, (short)0, acc, false, false);
  }

  const float* bp = gate == 0 ? b0 : gate == 1 ? b1 : gate == 2 ? b2 : b3;
  int n = lane & 15, hi = lane >> 4;
  float bias = bp[col + n];
  float* Gg = G + (size_t)gate * GS;
#pragma unroll
  for (int v = 0; v < 8; ++v) {
    int row = v + 8 * hi;                       // batch index
    Gg[((size_t)t * BATCH + row) * HID + col + n] = acc[v] + bias;
  }
}

// ---- phase 2: persistent recurrent scan ------------------------------------
// 64 blocks x 128 threads (4 waves). Block owns 16 h-columns; wave = gate.
// U (bf16, 8MB total) stays L2-resident; c lives in LDS; h double-buffered
// in global bf16 with one grid barrier per timestep.
__global__ void qlstm_scan_kernel(const bf16_t* __restrict__ qtu,
                                  const float* __restrict__ G,
                                  bf16_t* __restrict__ hbuf,   // [2][16][1024]
                                  float* __restrict__ out,     // [T][B][H]
                                  unsigned* __restrict__ bar) {
  __shared__ float zlds[4][BATCH][16];
  __shared__ float clds[BATCH][16];
  int lane = threadIdx.x & 31;
  int gate = threadIdx.x >> 5;
  int col0 = blockIdx.x << 4;
  int n = lane & 15, hi = lane >> 4;

  for (int i = threadIdx.x; i < BATCH * 16; i += blockDim.x)
    clds[i >> 4][i & 15] = 0.f;
  __syncthreads();

  const bf16_t* Bm = qtu + ((size_t)gate << 20);

  for (int t = 0; t < T_STEPS; ++t) {
    const bf16_t* hread  = hbuf + ((t & 1)       ? (size_t)(BATCH * HID) : 0);
    bf16_t*       hwrite = hbuf + (((t + 1) & 1) ? (size_t)(BATCH * HID) : 0);

    v8f acc = {};
#pragma unroll 4
    for (int k0 = 0; k0 < HID; k0 += 32) {
      v16bf a = load_a_frag(hread, HID, k0, lane);
      v16bf b = load_b_frag(Bm, HID, col0, k0, lane);
      acc = __builtin_amdgcn_wmma_f32_16x16x32_bf16(false, a, false, b, (short)0, acc, false, false);
    }
#pragma unroll
    for (int v = 0; v < 8; ++v) zlds[gate][v + 8 * hi][n] = acc[v];
    __syncthreads();

    for (int idx = threadIdx.x; idx < BATCH * 16; idx += blockDim.x) {
      int row = idx >> 4, c = idx & 15;
      int colg = col0 + c;
      size_t gidx = ((size_t)t * BATCH + row) * HID + colg;
      float zf = G[0 * GS + gidx] + zlds[0][row][c];
      float zi = G[1 * GS + gidx] + zlds[1][row][c];
      float zo = G[2 * GS + gidx] + zlds[2][row][c];
      float zc = G[3 * GS + gidx] + zlds[3][row][c];
      float ft = fsigmoid(zf), it = fsigmoid(zi), ot = fsigmoid(zo);
      float cnew = it * ftanh(zc) * 0.8f + ft * clds[row][c];  // drop = 1-0.2
      clds[row][c] = cnew;
      float h = ot * ftanh(cnew);
      out[gidx] = h;
      hwrite[(size_t)row * HID + colg] = (bf16_t)h;
    }

    // -------- grid-wide barrier (sense via generation counter) -------------
    __syncthreads();
    if (threadIdx.x == 0) {
      __threadfence();
      unsigned arrived = atomicAdd(&bar[0], 1u);
      if (arrived == NBLK_SCAN - 1) {
        bar[0] = 0u;
        __threadfence();
        atomicExch(&bar[1], (unsigned)(t + 1));
      } else {
        while (atomicAdd(&bar[1], 0u) < (unsigned)(t + 1))
          __builtin_amdgcn_s_sleep(2);
        __threadfence();
      }
    }
    __syncthreads();
  }
}

// ---------------------------------------------------------------------------
extern "C" void kernel_launch(void* const* d_in, const int* in_sizes, int n_in,
                              void* d_out, int out_size, void* d_ws, size_t ws_size,
                              hipStream_t stream) {
  const float* x = (const float*)d_in[0];
  const float* W[4] = {(const float*)d_in[1], (const float*)d_in[2],
                       (const float*)d_in[3], (const float*)d_in[4]};
  const float* bia[4] = {(const float*)d_in[5], (const float*)d_in[6],
                         (const float*)d_in[7], (const float*)d_in[8]};
  const float* U[4] = {(const float*)d_in[9], (const float*)d_in[10],
                       (const float*)d_in[11], (const float*)d_in[12]};

  char* ws = (char*)d_ws;
  unsigned* bar  = (unsigned*)ws;                               // 256 B
  bf16_t* hbuf   = (bf16_t*)(ws + 256);                         // 64 KB (2x h)
  bf16_t* xb     = (bf16_t*)(ws + 65792);                       // 49,152,000 B
  bf16_t* qtw    = (bf16_t*)(ws + 49217792);                    // 8,388,608 B
  bf16_t* qtu    = (bf16_t*)(ws + 57606400);                    // 8,388,608 B
  float*  G      = (float*)(ws + 65995008);                     // 393,216,000 B

  init_kernel<<<128, 256, 0, stream>>>(hbuf, bar);

  int nx = T_STEPS * BATCH * DIM;
  cvt_bf16_kernel<<<(nx + 255) / 256, 256, 0, stream>>>(x, xb, nx);

  for (int g = 0; g < 4; ++g) {
    qexpand_kernel<<<4096, 256, 0, stream>>>(W[g], qtw + ((size_t)g << 20));
    qexpand_kernel<<<4096, 256, 0, stream>>>(U[g], qtu + ((size_t)g << 20));
  }

  gemm_gates_kernel<<<dim3(T_STEPS, 64), 128, 0, stream>>>(
      xb, qtw, bia[0], bia[1], bia[2], bia[3], G);

  qlstm_scan_kernel<<<NBLK_SCAN, 128, 0, stream>>>(
      qtu, G, hbuf, (float*)d_out, bar);
}
